// FullNetInternals_63883343560754
// MI455X (gfx1250) — compile-verified, matrix operand order checked by reading, not run on hardware
//
#include <hip/hip_runtime.h>

#define NCOL 1024
#define LAY  60
#define NC   42
#define NK   8
#define HOD  64
#define CK   336   // NC*NK
#define HSC  16
#define NTIL 21    // CK/16 output tiles
#define SOLC 1361.0f
#define EPS_ 1e-6f

typedef __attribute__((ext_vector_type(16))) _Float16 v16h;
typedef __attribute__((ext_vector_type(8)))  float    v8f;

// w2 fragment store in d_ws: [NTIL*2 frags][32 lanes][16 halves], hi then lo
#define FRAGS   (NTIL * 2)
#define FRAG_HALVES (FRAGS * 32 * 16)

struct Smem {
  float h[64 * 64];                    // hidden activations (rows 60..63 zero-padded)
  float td[LAY * NC], sd[LAY * NC], rd[LAY * NC], Td[LAY * NC], Rd[LAY * NC];
  float rbdir[LAY * NC], rbdif[LAY * NC];
  float w1[HOD * 2], b1[HOD], b2[CK];
  float csplit[NC];
  float fdd[LAY + 1], fdf[LAY + 1], fup[LAY + 1], fab[LAY];
  float sdir[LAY], sdif[LAY], tdt[LAY], tdf[LAY];
  float rtoa;
};

__device__ __forceinline__ float softplusf(float x) {
  return fmaxf(x, 0.f) + log1pf(expf(-fabsf(x)));
}

// K index inside a 32-chunk for half `hi` (0..15) per the ISA 16-bit A/B layout:
// lanes 0-15 hold K 0..7 (halves 0..7) and K 16..23 (halves 8..15); lanes 16-31 are +8.
__device__ __forceinline__ int kmap(int hi, int laneHi) {
  return (hi < 8 ? hi : hi + 8) + (laneHi ? 8 : 0);
}

// e-split MLP: weights via uniform global pointers -> scalar loads (SGPR-resident)
__device__ __forceinline__ void esplit(float tau, float mu,
                                       const float* __restrict__ sw1,
                                       const float* __restrict__ sb1,
                                       const float* __restrict__ sw2,
                                       const float* __restrict__ sb2,
                                       float& e0, float& e1) {
  float o0 = sb2[0], o1 = sb2[1], o2 = sb2[2];
  #pragma unroll
  for (int j = 0; j < HSC; ++j) {
    float hh = tau * sw1[j * 2 + 0] + mu * sw1[j * 2 + 1] + sb1[j];
    hh = fmaxf(hh, 0.f);
    o0 += sw2[0 * HSC + j] * hh;
    o1 += sw2[1 * HSC + j] * hh;
    o2 += sw2[2 * HSC + j] * hh;
  }
  float m = fmaxf(o0, fmaxf(o1, o2));
  float x0 = expf(o0 - m), x1 = expf(o1 - m), x2 = expf(o2 - m);
  float inv = 1.f / (x0 + x1 + x2);
  e0 = x0 * inv; e1 = x1 * inv;
}

// ---- prep kernel: split od_w2 into hi/lo f16 WMMA B fragments (shared by all blocks)
__global__ __launch_bounds__(32) void prep_w2_frags(const float* __restrict__ od_w2,
                                                    _Float16* __restrict__ whi,
                                                    _Float16* __restrict__ wlo) {
  const int f    = blockIdx.x;          // t*2 + kc
  const int t    = f >> 1;
  const int kc   = f & 1;
  const int lane = threadIdx.x;
  const int laneHi = (lane >= 16) ? 1 : 0;
  const int lmod   = lane & 15;
  #pragma unroll
  for (int i = 0; i < 16; ++i) {
    int k = kc * 32 + kmap(i, laneHi);
    float w = od_w2[(size_t)(16 * t + lmod) * HOD + k];
    _Float16 hv = (_Float16)w;
    whi[((size_t)f * 32 + lane) * 16 + i] = hv;
    wlo[((size_t)f * 32 + lane) * 16 + i] = (_Float16)(w - (float)hv);
  }
}

__global__ __launch_bounds__(128) void radnet_kernel(
    const float* __restrict__ x_layers, const float* __restrict__ x_surface,
    const float* __restrict__ w_mu,     const float* __restrict__ w_spec,
    const float* __restrict__ od_w1,    const float* __restrict__ od_b1,
    const float* __restrict__ od_b2,
    const float* __restrict__ sc_w1,    const float* __restrict__ sc_b1,
    const float* __restrict__ sc_w2,    const float* __restrict__ sc_b2,
    const _Float16* __restrict__ wfrag_hi, const _Float16* __restrict__ wfrag_lo,
    float* __restrict__ out)
{
  extern __shared__ char smraw[];
  Smem* S = reinterpret_cast<Smem*>(smraw);
  const int n    = blockIdx.x;
  const int tid  = threadIdx.x;
  const int lane = tid & 31;
  const int wv   = tid >> 5;           // 4 waves of 32

  // ---------- phase 0: stage small params, zero accumulators ----------
  for (int i = tid; i < HOD * 2; i += 128) S->w1[i]  = od_w1[i];
  for (int i = tid; i < HOD;     i += 128) S->b1[i]  = od_b1[i];
  for (int i = tid; i < CK;      i += 128) S->b2[i]  = od_b2[i];
  if (tid == 0) {
    float mx = -1e30f;
    for (int c = 0; c < NC; ++c) mx = fmaxf(mx, w_spec[c]);
    float sum = 0.f;
    for (int c = 0; c < NC; ++c) { float e = expf(w_spec[c] - mx); S->csplit[c] = e; sum += e; }
    float inv = 1.f / sum;
    for (int c = 0; c < NC; ++c) S->csplit[c] *= inv;
    S->rtoa = 0.f;
  }
  for (int i = tid; i < LAY + 1; i += 128) { S->fdd[i] = 0.f; S->fdf[i] = 0.f; S->fup[i] = 0.f; }
  for (int i = tid; i < LAY;     i += 128) { S->fab[i] = 0.f; S->sdir[i] = 0.f; S->sdif[i] = 0.f;
                                             S->tdt[i] = 0.f; S->tdf[i] = 0.f; }
  __syncthreads();

  // ---------- phase 1: hidden layer h = softplus(tp @ od_w1^T + b1) ----------
  {
    int r  = tid >> 1;
    int k0 = (tid & 1) * 32;
    float tp0 = 0.f, tp1 = 0.f;
    if (r < LAY) {
      const float* p = x_layers + (size_t)(n * LAY + r) * 10;
      tp0 = p[0]; tp1 = p[1];
    }
    for (int i = 0; i < 32; ++i) {
      int k = k0 + i;
      float v = 0.f;
      if (r < LAY) v = softplusf(tp0 * S->w1[k * 2 + 0] + tp1 * S->w1[k * 2 + 1] + S->b1[k]);
      S->h[r * 64 + k] = v;
    }
  }
  __syncthreads();

  // ---------- phase 2: WMMA GEMM (split-f16 for ~fp32) + optics per (l,c) ----------
  const float mu_dir = x_surface[n * 5 + 0];
  const float mu_dif = 1.f / (1.f + expf(-w_mu[0]));
  const int laneHi = (lane >= 16) ? 1 : 0;
  const int lmod   = lane & 15;

  // A fragments for this wave's 16 layer-rows, both K-chunks, hi/lo split
  v16h ahi[2], alo[2];
  #pragma unroll
  for (int kc = 0; kc < 2; ++kc) {
    #pragma unroll
    for (int i = 0; i < 16; ++i) {
      int k = kc * 32 + kmap(i, laneHi);
      float v = S->h[(wv * 16 + lmod) * 64 + k];
      _Float16 hv = (_Float16)v;
      ahi[kc][i] = hv;
      alo[kc][i] = (_Float16)(v - (float)hv);
    }
  }

  // per-lane con values: k = lane&7 fixed; 8 rows per lane
  const int kk = lane & 7;
  float conr[8];
  #pragma unroll
  for (int v = 0; v < 8; ++v) {
    int row = wv * 16 + v + (laneHi ? 8 : 0);
    conr[v] = (row < LAY) ? x_layers[(size_t)(n * LAY + row) * 10 + 2 + kk] : 0.f;
  }

  const v16h* __restrict__ bhiF = reinterpret_cast<const v16h*>(wfrag_hi);
  const v16h* __restrict__ bloF = reinterpret_cast<const v16h*>(wfrag_lo);

  for (int t = 0; t < NTIL; ++t) {
    const int o = 16 * t + lmod;        // output index 0..335, fixed per lane
    v8f acc = {0.f, 0.f, 0.f, 0.f, 0.f, 0.f, 0.f, 0.f};
    #pragma unroll
    for (int kc = 0; kc < 2; ++kc) {
      v16h bhi = bhiF[(size_t)(t * 2 + kc) * 32 + lane];
      v16h blo = bloF[(size_t)(t * 2 + kc) * 32 + lane];
      acc = __builtin_amdgcn_wmma_f32_16x16x32_f16(false, ahi[kc], false, bhi, (short)0, acc, false, false);
      acc = __builtin_amdgcn_wmma_f32_16x16x32_f16(false, ahi[kc], false, blo, (short)0, acc, false, false);
      acc = __builtin_amdgcn_wmma_f32_16x16x32_f16(false, alo[kc], false, bhi, (short)0, acc, false, false);
    }

    // sigma = softplus(acc + b2[o]); tau partial = sigma * con[k]
    const float bias = S->b2[o];
    float taup[8];
    #pragma unroll
    for (int v = 0; v < 8; ++v) taup[v] = softplusf(acc[v] + bias) * conr[v];

    // reduce over k = lane&7 (butterfly within groups of 8 lanes)
    #pragma unroll
    for (int m = 1; m < 8; m <<= 1)
      #pragma unroll
      for (int v = 0; v < 8; ++v)
        taup[v] += __shfl_xor(taup[v], m, 32);

    // redistribute: lane now owns (m_rel = lane&15, cc = lane>>4)
    const int m_rel = lmod, cc = laneHi;
    const int src = ((m_rel & 8) ? 16 : 0) + 8 * cc;
    float tau = 0.f;
    #pragma unroll
    for (int v = 0; v < 8; ++v) {
      float tm = __shfl(taup[v], src, 32);
      if ((m_rel & 7) == v) tau = tm;
    }

    const int l = wv * 16 + m_rel;
    const int c = 2 * t + cc;
    if (l < LAY) {
      float tdv  = expf(-tau / (mu_dir + EPS_));
      float tdfv = expf(-tau / (mu_dif + EPS_));
      float e0d, e1d, e0f, e1f;
      esplit(tau, mu_dir, sc_w1, sc_b1, sc_w2, sc_b2, e0d, e1d);
      esplit(tau, mu_dif, sc_w1, sc_b1, sc_w2, sc_b2, e0f, e1f);
      float sdv = (1.f - tdv) * e0d;
      float rdv = (1.f - tdv) * e1d;
      float Tdv = tdfv + (1.f - tdfv) * e0f;
      float Rdv = (1.f - tdfv) * e1f;
      int idx = l * NC + c;
      S->td[idx] = tdv; S->sd[idx] = sdv; S->rd[idx] = rdv;
      S->Td[idx] = Tdv; S->Rd[idx] = Rdv;
      float cs = S->csplit[c];
      atomicAdd(&S->sdir[l], (1.f - tdv)  * (e0d + e1d) * cs);
      atomicAdd(&S->sdif[l], (1.f - tdfv) * (e0f + e1f) * cs);
      atomicAdd(&S->tdt[l],  tdv  * cs);
      atomicAdd(&S->tdf[l],  tdfv * cs);
    }
  }
  __syncthreads();

  // ---------- phase 3: adding-doubling scans, one thread per channel ----------
  if (tid < NC) {
    const int c = tid;
    const float cs    = S->csplit[c];
    const float rsdir = x_surface[n * 5 + 1];
    const float rsdif = x_surface[n * 5 + 2];

    // reverse scan: reflectance build-up from surface
    float rbd = rsdir, rbf = rsdif;
    for (int l = LAY - 1; l >= 0; --l) {
      int idx = l * NC + c;
      float tdv = S->td[idx], sdv = S->sd[idx], rdv = S->rd[idx];
      float Tdv = S->Td[idx], Rdv = S->Rd[idx];
      S->rbdir[idx] = rbd;
      S->rbdif[idx] = rbf;
      float d = 1.f / (1.f - Rdv * rbf);
      float radir = rdv + Tdv * d * (tdv * rbd + sdv * rbf);
      float radif = Rdv + Tdv * Tdv * d * rbf;
      rbd = radir; rbf = radif;
    }
    atomicAdd(&S->rtoa, rbd * cs);

    // forward scan: flux propagation
    float fdir = cs * mu_dir * SOLC, fdif = 0.f;
    atomicAdd(&S->fdd[0], fdir);
    for (int l = 0; l < LAY; ++l) {
      int idx = l * NC + c;
      float tdv = S->td[idx], sdv = S->sd[idx], rdv = S->rd[idx];
      float Tdv = S->Td[idx], Rdv = S->Rd[idx];
      float rb_d = S->rbdir[idx], rb_f = S->rbdif[idx];
      float d = 1.f / (1.f - Rdv * rb_f);
      float a = tdv * fdir;
      float D = d * (Tdv * fdif + sdv * fdir + Rdv * rb_d * a);
      float U = rb_d * a + rb_f * D;
      float up = rdv * fdir + Rdv * fdif + Tdv * U;
      float absb = fdir + fdif + U - a - D - up;
      atomicAdd(&S->fdd[l + 1], a);
      atomicAdd(&S->fdf[l + 1], D);
      atomicAdd(&S->fup[l], up);
      atomicAdd(&S->fab[l], absb);
      fdir = a; fdif = D;
    }
    atomicAdd(&S->fup[LAY], rsdir * fdir + rsdif * fdif);
  }
  __syncthreads();

  // ---------- phase 4: write outputs (each element exactly once) ----------
  float* fdd_o  = out;                                   // (N, L+1)
  float* fdf_o  = fdd_o  + (size_t)NCOL * (LAY + 1);     // (N, L+1)
  float* fup_o  = fdf_o  + (size_t)NCOL * (LAY + 1);     // (N, L+1)
  float* fab_o  = fup_o  + (size_t)NCOL * (LAY + 1);     // (N, L)
  float* rtoa_o = fab_o  + (size_t)NCOL * LAY;           // (N,)
  float* sdir_o = rtoa_o + NCOL;                         // (N, L)
  float* sdif_o = sdir_o + (size_t)NCOL * LAY;           // (N, L)
  float* tdt_o  = sdif_o + (size_t)NCOL * LAY;           // (N, L)
  float* tdf_o  = tdt_o  + (size_t)NCOL * LAY;           // (N, L)

  if (tid < LAY + 1) {
    fdd_o[(size_t)n * (LAY + 1) + tid] = S->fdd[tid];
    fdf_o[(size_t)n * (LAY + 1) + tid] = S->fdf[tid];
    fup_o[(size_t)n * (LAY + 1) + tid] = S->fup[tid];
  }
  if (tid < LAY) {
    fab_o [(size_t)n * LAY + tid] = S->fab[tid];
    sdir_o[(size_t)n * LAY + tid] = S->sdir[tid];
    sdif_o[(size_t)n * LAY + tid] = S->sdif[tid];
    tdt_o [(size_t)n * LAY + tid] = S->tdt[tid];
    tdf_o [(size_t)n * LAY + tid] = S->tdf[tid];
  }
  if (tid == 0) rtoa_o[n] = S->rtoa;
}

extern "C" void kernel_launch(void* const* d_in, const int* in_sizes, int n_in,
                              void* d_out, int out_size, void* d_ws, size_t ws_size,
                              hipStream_t stream) {
  (void)in_sizes; (void)n_in; (void)out_size; (void)ws_size;
  _Float16* whi = reinterpret_cast<_Float16*>(d_ws);
  _Float16* wlo = whi + FRAG_HALVES;

  // build L2-resident pre-swizzled hi/lo f16 B fragments of od_w2
  prep_w2_frags<<<dim3(FRAGS), dim3(32), 0, stream>>>(
      (const float*)d_in[6], whi, wlo);

  hipFuncSetAttribute(reinterpret_cast<const void*>(radnet_kernel),
                      hipFuncAttributeMaxDynamicSharedMemorySize,
                      (int)sizeof(Smem));
  radnet_kernel<<<dim3(NCOL), dim3(128), sizeof(Smem), stream>>>(
      (const float*)d_in[0],  (const float*)d_in[1],
      (const float*)d_in[2],  (const float*)d_in[3],
      (const float*)d_in[4],  (const float*)d_in[5],
      (const float*)d_in[7],
      (const float*)d_in[8],  (const float*)d_in[9],
      (const float*)d_in[10], (const float*)d_in[11],
      (const _Float16*)whi,   (const _Float16*)wlo,
      (float*)d_out);
}
